// Decoder_11287174054404
// MI455X (gfx1250) — matrix operand order repaired and checked by reading, not compiled, not run on hardware
//
#include <hip/hip_runtime.h>
#include <hip/hip_bf16.h>

// ---------------------------------------------------------------------------
// CDNA5 (gfx1250, wave32) implementation of the quantized decoder.
// All heavy convs (3x3 64->64, 1x1 32->32, 9x9 16->3) run as implicit GEMM
// on v_wmma_f32_16x16x32_f16 (operands are 6-bit fake-quantized -> f16 exact,
// f32 accumulate). f16 conv inputs are kept NHWC so A-fragments load as two
// 16-byte ds_load_b128 / global_load_b128 per lane. The 9x9 conv pads
// K = 81 taps * 16ch = 1296 to 41x32 with a zeroed tap so the K-loop is
// branch-free. Everything else is bandwidth-bound scalar code.
// ---------------------------------------------------------------------------

typedef _Float16 half_t;
typedef __attribute__((ext_vector_type(16))) _Float16 v16h;
typedef __attribute__((ext_vector_type(8)))  _Float16 v8h;
typedef __attribute__((ext_vector_type(8)))  float    v8f;

#define QMAXF 31.0f

__device__ __forceinline__ float scale_of(const unsigned* amax) {
    float a = __uint_as_float(*amax);
    return fmaxf(a, 1e-8f) / QMAXF;
}
__device__ __forceinline__ float dq(float v, float s) {
    float q = rintf(v / s);                 // round-to-nearest-even, like jnp.round
    q = fminf(fmaxf(q, -QMAXF), QMAXF);
    return q * s;
}
__device__ __forceinline__ int refl(int x, int n) {
    if (x < 0) x = -x;
    if (x >= n) x = 2 * n - 2 - x;
    return x;
}

// --------------------------- utility kernels -------------------------------

__global__ void k_zero_u32(unsigned* p, int n) {
    int i = blockIdx.x * blockDim.x + threadIdx.x;
    if (i < n) p[i] = 0u;
}

__global__ void k_absmax(const float* __restrict__ x, size_t n, unsigned* amax) {
    __shared__ float sm[256];
    float m = 0.f;
    for (size_t i = blockIdx.x * (size_t)blockDim.x + threadIdx.x; i < n;
         i += (size_t)gridDim.x * blockDim.x)
        m = fmaxf(m, fabsf(x[i]));
    sm[threadIdx.x] = m;
    __syncthreads();
    for (int s = 128; s > 0; s >>= 1) {
        if (threadIdx.x < s) sm[threadIdx.x] = fmaxf(sm[threadIdx.x], sm[threadIdx.x + s]);
        __syncthreads();
    }
    if (threadIdx.x == 0) atomicMax(amax, __float_as_uint(sm[0]));
}

__global__ void k_quant_f32(const float* __restrict__ x, const unsigned* __restrict__ amax,
                            float* __restrict__ y, size_t n) {
    const float s = scale_of(amax);
    for (size_t i = blockIdx.x * (size_t)blockDim.x + threadIdx.x; i < n;
         i += (size_t)gridDim.x * blockDim.x)
        y[i] = dq(x[i], s);
}

__global__ void k_quant_f16(const float* __restrict__ x, const unsigned* __restrict__ amax,
                            half_t* __restrict__ y, size_t n) {
    const float s = scale_of(amax);
    for (size_t i = blockIdx.x * (size_t)blockDim.x + threadIdx.x; i < n;
         i += (size_t)gridDim.x * blockDim.x)
        y[i] = (half_t)dq(x[i], s);
}

// NCHW f32 in -> quantized f32 NCHW + quantized f16 NHWC
__global__ void k_quant_both_nhwc(const float* __restrict__ x,
                                  const unsigned* __restrict__ amax,
                                  float* __restrict__ yf, half_t* __restrict__ yh,
                                  size_t n, int C, int HW) {
    const float s = scale_of(amax);
    for (size_t i = blockIdx.x * (size_t)blockDim.x + threadIdx.x; i < n;
         i += (size_t)gridDim.x * blockDim.x) {
        float q = dq(x[i], s);
        yf[i] = q;
        size_t pix = i % HW;
        size_t t = i / HW;
        int c = (int)(t % C);
        size_t nn = t / C;
        yh[(nn * HW + pix) * C + c] = (half_t)q;
    }
}

// NCHW f32 in -> quantized f16 NHWC
__global__ void k_quant_f16_nhwc(const float* __restrict__ x,
                                 const unsigned* __restrict__ amax,
                                 half_t* __restrict__ y, size_t n, int C, int HW) {
    const float s = scale_of(amax);
    for (size_t i = blockIdx.x * (size_t)blockDim.x + threadIdx.x; i < n;
         i += (size_t)gridDim.x * blockDim.x) {
        size_t pix = i % HW;
        size_t t = i / HW;
        int c = (int)(t % C);
        size_t nn = t / C;
        y[(nn * HW + pix) * C + c] = (half_t)dq(x[i], s);
    }
}

// per-(n,c) mean of the quantized tensor; blockIdx.x = n*C + c
__global__ void k_meanq(const float* __restrict__ x, const unsigned* __restrict__ amax,
                        float* __restrict__ mean, int HW) {
    const float s = scale_of(amax);
    const float* xp = x + (size_t)blockIdx.x * HW;
    __shared__ float sm[256];
    float acc = 0.f;
    for (int i = threadIdx.x; i < HW; i += 256) acc += dq(xp[i], s);
    sm[threadIdx.x] = acc;
    __syncthreads();
    for (int t = 128; t > 0; t >>= 1) {
        if (threadIdx.x < t) sm[threadIdx.x] += sm[threadIdx.x + t];
        __syncthreads();
    }
    if (threadIdx.x == 0) mean[blockIdx.x] = sm[0] / (float)HW;
}

__global__ void k_q_submean(const float* __restrict__ x, const unsigned* __restrict__ amax,
                            const float* __restrict__ mean, float* __restrict__ y,
                            size_t n, int HW) {
    const float s = scale_of(amax);
    for (size_t i = blockIdx.x * (size_t)blockDim.x + threadIdx.x; i < n;
         i += (size_t)gridDim.x * blockDim.x)
        y[i] = dq(x[i], s) - mean[i / HW];
}

__global__ void k_quant_addmean(const float* __restrict__ x, const unsigned* __restrict__ amax,
                                const float* __restrict__ mean, float* __restrict__ y,
                                size_t n, int HW) {
    const float s = scale_of(amax);
    for (size_t i = blockIdx.x * (size_t)blockDim.x + threadIdx.x; i < n;
         i += (size_t)gridDim.x * blockDim.x)
        y[i] = dq(x[i], s) + mean[i / HW];
}

__global__ void k_mod_mul(const float* __restrict__ x, const float* __restrict__ mod,
                          float* __restrict__ y, size_t n, int HW) {
    for (size_t i = blockIdx.x * (size_t)blockDim.x + threadIdx.x; i < n;
         i += (size_t)gridDim.x * blockDim.x)
        y[i] = mod[i / HW] * x[i];
}

__global__ void k_add(const float* __restrict__ a, const float* __restrict__ b,
                      float* __restrict__ y, size_t n) {
    for (size_t i = blockIdx.x * (size_t)blockDim.x + threadIdx.x; i < n;
         i += (size_t)gridDim.x * blockDim.x)
        y[i] = a[i] + b[i];
}

__global__ void k_add_relu(const float* __restrict__ a, const float* __restrict__ b,
                           float* __restrict__ y, size_t n) {
    for (size_t i = blockIdx.x * (size_t)blockDim.x + threadIdx.x; i < n;
         i += (size_t)gridDim.x * blockDim.x)
        y[i] = fmaxf(a[i] + b[i], 0.f);
}

__global__ void k_relu(const float* __restrict__ x, float* __restrict__ y, size_t n) {
    for (size_t i = blockIdx.x * (size_t)blockDim.x + threadIdx.x; i < n;
         i += (size_t)gridDim.x * blockDim.x)
        y[i] = fmaxf(x[i], 0.f);
}

// quantize + 2x nearest upsample + f16 convert; in NCHW f32, out NHWC f16
__global__ void k_up2_q_f16(const float* __restrict__ x, const unsigned* __restrict__ amax,
                            half_t* __restrict__ y, int Nn, int C, int H, int W) {
    const float s = scale_of(amax);
    const int W2 = 2 * W, H2 = 2 * H;
    const size_t total = (size_t)Nn * C * H2 * W2;
    for (size_t i = blockIdx.x * (size_t)blockDim.x + threadIdx.x; i < total;
         i += (size_t)gridDim.x * blockDim.x) {
        int c = (int)(i % C);            // NHWC linear: c innermost
        size_t t = i / C;
        int ow = (int)(t % W2); t /= W2;
        int oh = (int)(t % H2);
        int n = (int)(t / H2);
        float v = x[(((size_t)n * C + c) * H + (oh >> 1)) * W + (ow >> 1)];
        y[i] = (half_t)dq(v, s);
    }
}

// ----------------------- WMMA weight packing -------------------------------
// B-fragment layout for v_wmma_*_16x16x32_f16 (wave32, ISA 7.12.2):
//   lane l holds column n = l&15; half j holds K = (l>>4)*8 + (j&7) + (j>>3)*16.

__global__ void k_pack3x3(const float* __restrict__ w, const unsigned* __restrict__ amax,
                          half_t* __restrict__ wp, int Cin) {
    int idx = blockIdx.x * blockDim.x + threadIdx.x;
    if (idx >= 9 * 2 * 4 * 32 * 16) return;
    int j     = idx & 15;
    int lane  = (idx >> 4) & 31;
    int g     = (idx >> 9) & 3;
    int chalf = (idx >> 11) & 1;
    int tap   = idx >> 12;
    int cin   = chalf * 32 + (lane >> 4) * 8 + (j & 7) + (j >> 3) * 16;
    int cout  = g * 16 + (lane & 15);
    int ky = tap / 3, kx = tap % 3;
    float s = scale_of(amax);
    float v = w[(((size_t)cout * Cin + cin) * 3 + ky) * 3 + kx];
    wp[idx] = (half_t)dq(v, s);
}

// 1x1 32->32: wp[g][lane][j], cin = (lane>>4)*8 + (j&7) + (j>>3)*16, cout = g*16 + (lane&15)
__global__ void k_pack1x1(const float* __restrict__ w, const unsigned* __restrict__ amax,
                          half_t* __restrict__ wp) {
    int idx = blockIdx.x * blockDim.x + threadIdx.x;
    if (idx >= 2 * 32 * 16) return;
    int j    = idx & 15;
    int lane = (idx >> 4) & 31;
    int g    = idx >> 9;
    int cin  = (lane >> 4) * 8 + (j & 7) + (j >> 3) * 16;
    int cout = g * 16 + (lane & 15);
    float s = scale_of(amax);
    wp[idx] = (half_t)dq(w[(size_t)cout * 32 + cin], s);
}

// 9x9 16->3: K = tap*16 + ci padded to 41*32; wp[ks][lane][j];
// Kg = ks*32 + (lane>>4)*8 + (j&7) + (j>>3)*16; tap = Kg/16 (zero for tap>=81), ci = Kg%16;
// cout = lane&15 (zero for cout>=3).
__global__ void k_pack9x9(const float* __restrict__ w, const unsigned* __restrict__ amax,
                          half_t* __restrict__ wp) {
    int idx = blockIdx.x * blockDim.x + threadIdx.x;
    if (idx >= 41 * 32 * 16) return;
    int j    = idx & 15;
    int lane = (idx >> 4) & 31;
    int ks   = idx >> 9;
    int Kg   = ks * 32 + (lane >> 4) * 8 + (j & 7) + (j >> 3) * 16;
    int tap  = Kg >> 4;
    int ci   = Kg & 15;
    int co   = lane & 15;
    float v = 0.f;
    if (tap < 81 && co < 3) {
        float s = scale_of(amax);
        int ky = tap / 9, kx = tap % 9;
        v = dq(w[(((size_t)co * 16 + ci) * 9 + ky) * 9 + kx], s);
    }
    wp[idx] = (half_t)v;
}

// -------------------- 3x3 conv 64->64, reflect pad, WMMA -------------------
// Block = 128 threads = 4 wave32. One 16-pixel x 64-cout tile per block.
__global__ __launch_bounds__(128)
void k_conv3x3_wmma(const half_t* __restrict__ x,    // [N,H,W,64] quantized f16
                    const half_t* __restrict__ wp,   // packed [9][2][4][32][16]
                    const float* __restrict__ bias,  // [64] raw
                    float* __restrict__ y,           // [N,64,H,W] f32
                    int Nn, int H, int W) {
    const int C = 64;
    const int lane = threadIdx.x & 31;
    const int wave = threadIdx.x >> 5;    // cout group 0..3
    const int tilesPerRow = W / 16;
    int t = blockIdx.x;
    const int wt = t % tilesPerRow; t /= tilesPerRow;
    const int h  = t % H;           t /= H;
    const int n  = t;
    const int w0 = wt * 16;

    __shared__ __align__(16) half_t patch[3][18][64];   // 3456 halves
    for (int chunk = threadIdx.x; chunk < 3 * 18 * 8; chunk += 128) {
        int c8  = chunk & 7;
        int pos = chunk >> 3;
        int col = pos % 18;
        int row = pos / 18;
        int hh = refl(h + row - 1, H);
        int ww = refl(w0 + col - 1, W);
        const half_t* src = x + (((size_t)n * H + hh) * W + ww) * C + c8 * 8;
        *(v8h*)&patch[row][col][c8 * 8] = *(const v8h*)src;
    }
    __syncthreads();

    v8f acc = {};
    const int m  = lane & 15;
    const int kb = (lane >> 4) * 8;
    for (int tap = 0; tap < 9; ++tap) {
        const int dy = tap / 3, dx = tap % 3;
        const half_t* wtap = wp + (size_t)tap * (2 * 4 * 32 * 16);
        if (tap + 1 < 9)
            __builtin_prefetch(wtap + 2 * 4 * 32 * 16, 0, 0);   // global_prefetch_b8
        for (int chalf = 0; chalf < 2; ++chalf) {
            const half_t* ap = &patch[dy][m + dx][chalf * 32 + kb];
            union { v16h v; v8h h8[2]; } au;
            au.h8[0] = *(const v8h*)ap;          // ds_load_b128
            au.h8[1] = *(const v8h*)(ap + 16);   // ds_load_b128
            const half_t* bp = wtap + (((size_t)chalf * 4 + wave) * 32 + lane) * 16;
            v16h b = *(const v16h*)bp;
            acc = __builtin_amdgcn_wmma_f32_16x16x32_f16(
                false, au.v, false, b, (short)0, acc, false, false);
        }
    }
    __builtin_amdgcn_s_wait_tensorcnt(0);  // CDNA5 marker; counter is always 0 here

    const int co = wave * 16 + (lane & 15);
    const float bv = bias[co];
#pragma unroll
    for (int r = 0; r < 8; ++r) {
        int mm = r + (lane >> 4) * 8;
        y[(((size_t)n * C + co) * H + h) * W + w0 + mm] = acc[r] + bv;
    }
}

// -------------------- 1x1 conv 32->32 (K=32) via WMMA, fused relu ----------
__global__ __launch_bounds__(64)
void k_conv1x1_wmma32(const half_t* __restrict__ x,   // [N*HW,32] NHWC f16
                      const half_t* __restrict__ wp,  // packed [2][32][16]
                      const float* __restrict__ bias, // [32]
                      float* __restrict__ y,          // [N,32,HW] f32 (relu)
                      int Nn, int HW) {
    const int lane = threadIdx.x & 31;
    const int wave = threadIdx.x >> 5;
    const size_t p0 = (size_t)blockIdx.x * 16;
    const int n = (int)(p0 / (size_t)HW);
    const size_t pix0 = p0 - (size_t)n * HW;
    const int m  = lane & 15;
    const int kb = (lane >> 4) * 8;
    const half_t* xp = x + (p0 + m) * 32;
    union { v16h v; v8h h8[2]; } au;
    au.h8[0] = *(const v8h*)(xp + kb);        // global_load_b128
    au.h8[1] = *(const v8h*)(xp + kb + 16);   // global_load_b128
    v16h b = *(const v16h*)(wp + ((size_t)wave * 32 + lane) * 16);
    v8f acc = {};
    acc = __builtin_amdgcn_wmma_f32_16x16x32_f16(false, au.v, false, b, (short)0, acc,
                                                 false, false);
    const int co = wave * 16 + m;
    const float bv = bias[co];
#pragma unroll
    for (int r = 0; r < 8; ++r) {
        int mm = r + (lane >> 4) * 8;
        y[((size_t)n * 32 + co) * HW + pix0 + mm] = fmaxf(acc[r] + bv, 0.f);
    }
}

// -------------------- 9x9 conv 16->3, reflect pad 4, WMMA ------------------
// Block = 128 threads = 4 wave32 over one 64-pixel row segment.
// K = 81 taps * 16ch = 1296, padded to 41*32 via zero tap 81 (patch row 9 is
// zeroed; packed weights zero there too) -> branch-free K-loop, A = two
// ds_load_b128 per WMMA (8-half runs never cross a tap boundary).
__global__ __launch_bounds__(128)
void k_conv9x9_wmma(const half_t* __restrict__ x,    // [N,H,W,16] NHWC f16
                    const half_t* __restrict__ wp,   // packed [41][32][16]
                    const float* __restrict__ bias,  // [3]
                    float* __restrict__ y,           // [N,3,H,W] f32
                    int Nn, int H, int W) {
    const int lane = threadIdx.x & 31;
    const int wave = threadIdx.x >> 5;
    const int segsPerRow = W / 64;
    int t = blockIdx.x;
    const int seg = t % segsPerRow; t /= segsPerRow;
    const int h   = t % H;          t /= H;
    const int n   = t;
    const int w0  = seg * 64;

    __shared__ __align__(16) half_t patch[10][72][16];   // 23040 B
    for (int chunk = threadIdx.x; chunk < 9 * 72 * 2; chunk += 128) {
        int c8  = chunk & 1;
        int pos = chunk >> 1;
        int col = pos % 72;
        int row = pos / 72;
        int hh = refl(h + row - 4, H);
        int ww = refl(w0 + col - 4, W);
        const half_t* src = x + (((size_t)n * H + hh) * W + ww) * 16 + c8 * 8;
        *(v8h*)&patch[row][col][c8 * 8] = *(const v8h*)src;
    }
    for (int chunk = threadIdx.x; chunk < 72 * 2; chunk += 128) {
        int c8 = chunk & 1, col = chunk >> 1;
        v8h z = {};
        *(v8h*)&patch[9][col][c8 * 8] = z;    // zero pad tap 81
    }
    __syncthreads();

    v8f acc = {};
    const int mA = (lane & 15) + wave * 16;   // pixel within 64-seg (A rows)
    const int kb = (lane >> 4) * 8;
    for (int ks = 0; ks < 41; ++ks) {
        const int r0   = ks * 32 + kb;        // 8-half run: one tap, ch cf0..cf0+7
        const int tap0 = r0 >> 4;
        const int cf0  = r0 & 15;
        const int tap1 = tap0 + 1;            // second run: next tap, same ch half
        union { v16h v; v8h h8[2]; } au;
        au.h8[0] = *(const v8h*)&patch[tap0 / 9][mA + tap0 % 9][cf0];
        au.h8[1] = *(const v8h*)&patch[tap1 / 9][mA + tap1 % 9][cf0];
        v16h b = *(const v16h*)(wp + ((size_t)ks * 32 + lane) * 16);
        acc = __builtin_amdgcn_wmma_f32_16x16x32_f16(false, au.v, false, b, (short)0,
                                                     acc, false, false);
    }
    const int co = lane & 15;
    if (co < 3) {
        const float bv = bias[co];
#pragma unroll
        for (int r = 0; r < 8; ++r) {
            int mm = r + (lane >> 4) * 8;
            int pix = w0 + wave * 16 + mm;
            y[(((size_t)n * 3 + co) * H + h) * W + pix] = acc[r] + bv;
        }
    }
}

// ---------- grouped 3x3 conv (2 cin per cout), NHWC in, fused relu ---------
__global__ void k_gconv3x3_relu(const half_t* __restrict__ x,   // [N,H,W,Cin]
                                const half_t* __restrict__ wq,
                                const float* __restrict__ bias,
                                float* __restrict__ y,          // [N,Cout,H,W]
                                int Nn, int Cout, int H, int W) {
    const int Cin = 2 * Cout;
    const size_t total = (size_t)Nn * Cout * H * W;
    for (size_t i = blockIdx.x * (size_t)blockDim.x + threadIdx.x; i < total;
         i += (size_t)gridDim.x * blockDim.x) {
        int w_ = (int)(i % W);
        size_t t = i / W;
        int h  = (int)(t % H); t /= H;
        int co = (int)(t % Cout);
        int n  = (int)(t / Cout);
        float acc = bias[co];
        for (int ky = 0; ky < 3; ++ky) {
            int hh = refl(h + ky - 1, H);
            for (int kx = 0; kx < 3; ++kx) {
                int ww = refl(w_ + kx - 1, W);
                const half_t* xp = x + (((size_t)n * H + hh) * W + ww) * Cin + co * 2;
                acc += (float)xp[0] * (float)wq[((size_t)co * 2 + 0) * 9 + ky * 3 + kx];
                acc += (float)xp[1] * (float)wq[((size_t)co * 2 + 1) * 9 + ky * 3 + kx];
            }
        }
        y[i] = fmaxf(acc, 0.f);
    }
}

// ---------- 1x1 conv 16->16 scalar, NHWC in, fused relu --------------------
__global__ void k_conv1x1_16_relu(const half_t* __restrict__ x,  // [N*HW,16]
                                  const half_t* __restrict__ wq,
                                  const float* __restrict__ bias,
                                  float* __restrict__ y,         // [N,16,HW]
                                  int Nn, int C, int HW) {
    const size_t total = (size_t)Nn * C * HW;
    for (size_t i = blockIdx.x * (size_t)blockDim.x + threadIdx.x; i < total;
         i += (size_t)gridDim.x * blockDim.x) {
        size_t pix = i % HW;
        size_t t = i / HW;
        int co = (int)(t % C);
        int n  = (int)(t / C);
        const half_t* xp = x + ((size_t)n * HW + pix) * C;
        float acc = bias[co];
        for (int ci = 0; ci < 16; ++ci)
            acc += (float)xp[ci] * (float)wq[co * C + ci];
        y[i] = fmaxf(acc, 0.f);
    }
}

// ===========================================================================
extern "C" void kernel_launch(void* const* d_in, const int* in_sizes, int n_in,
                              void* d_out, int out_size, void* d_ws, size_t ws_size,
                              hipStream_t stream) {
    const float* content1 = (const float*)d_in[0];
    const float* style0   = (const float*)d_in[1];
    const float* style1   = (const float*)d_in[2];
    const float* weight0  = (const float*)d_in[3];
    const float* weight1  = (const float*)d_in[4];
    const float* bias0    = (const float*)d_in[5];
    const float* bias1    = (const float*)d_in[6];
    const float* w_d1_1 = (const float*)d_in[7];  const float* b_d1_1 = (const float*)d_in[8];
    const float* w_d1_2 = (const float*)d_in[9];  const float* b_d1_2 = (const float*)d_in[10];
    const float* w_d2_1 = (const float*)d_in[11]; const float* b_d2_1 = (const float*)d_in[12];
    const float* w_d2_2 = (const float*)d_in[13]; const float* b_d2_2 = (const float*)d_in[14];
    const float* w31 = (const float*)d_in[15]; const float* b31 = (const float*)d_in[16];
    const float* w32 = (const float*)d_in[17]; const float* b32 = (const float*)d_in[18];
    const float* w33 = (const float*)d_in[19]; const float* b33 = (const float*)d_in[20];
    const float* w34 = (const float*)d_in[21]; const float* b34 = (const float*)d_in[22];
    const float* w35 = (const float*)d_in[23]; const float* b35 = (const float*)d_in[24];

    const int NB = 4, CH = 64, HH = 128, WW = 128;
    const int HW1 = HH * WW;
    const size_t P1 = (size_t)NB * CH * HW1;            // 4,194,304
    const size_t P2 = (size_t)NB * 32 * 256 * 256;      // 8,388,608
    const size_t P3 = (size_t)NB * 16 * 512 * 512;      // 16,777,216
    const size_t PO = (size_t)NB * 3 * 512 * 512;       // 3,145,728

    // ---- small scratch region ----
    unsigned* SC = (unsigned*)d_ws;                     // 128 scale slots
    char* base = (char*)d_ws;
    float* mean_s1 = (float*)(base + 1024);
    float* mean_s0 = (float*)(base + 2048);
    float* mean_c1 = (float*)(base + 3072);
    float* mean_x2 = (float*)(base + 4096);
    float* qw1 = (float*)(base + 5120);
    float* qb1 = (float*)(base + 6144);
    float* qw0 = (float*)(base + 7168);
    float* qb0 = (float*)(base + 8192);
    half_t* WP11 = (half_t*)(base + 16384);             // 73728 B each
    half_t* WP12 = (half_t*)(base + 16384 + 73728);
    half_t* WP21 = (half_t*)(base + 16384 + 2 * 73728);
    half_t* WP22 = (half_t*)(base + 16384 + 3 * 73728);
    half_t* WP32 = (half_t*)(base + 16384 + 4 * 73728);           // 2 KB
    half_t* W31Q = (half_t*)(base + 16384 + 4 * 73728 + 2048);
    half_t* W33Q = (half_t*)(base + 16384 + 4 * 73728 + 4096);
    half_t* W34Q = (half_t*)(base + 16384 + 4 * 73728 + 6144);
    half_t* WPX  = (half_t*)(base + 393216);            // 41*32*16*2 = 41984 B

    // ---- big arena: 16 MB pages starting at 1 MB ----
    auto pg = [&](int p) -> char* { return base + (1u << 20) + (size_t)p * (16u << 20); };
    float*  X    = (float*)pg(0);
    float*  T1   = (float*)pg(1);
    float*  T2   = (float*)pg(2);
    float*  T3   = (float*)pg(3);
    float*  Q32  = (float*)pg(4);
    float*  RESO = (float*)pg(5);
    half_t* Q16  = (half_t*)pg(6);       // NHWC
    // dec3 overlays (earlier pages are dead by the time these are used)
    half_t* U64_16 = (half_t*)pg(8);     // 32 MB NHWC [4,256,256,64]
    float*  G31    = (float*)pg(10);     // 32 MB NCHW
    half_t* G31_16 = (half_t*)pg(6);     // 16 MB NHWC (reuse)
    float*  C32    = (float*)pg(0);      // 32 MB NCHW (reuse p0-p1)
    half_t* U32_16 = (half_t*)pg(2);     // 64 MB NHWC (p2-p5)
    float*  G33    = (float*)pg(8);      // 64 MB NCHW (p8-p11)
    half_t* G33_16 = (half_t*)pg(12);    // 32 MB NHWC
    float*  C34    = (float*)pg(0);      // 64 MB NCHW (p0-p3, dead by now)
    half_t* C34_16 = (half_t*)pg(4);     // 32 MB NHWC
    float*  FIN    = (float*)pg(8);      // 12 MB

    int sc = 0;
    auto absmax = [&](const float* x, size_t n, int slot) {
        k_absmax<<<1024, 256, 0, stream>>>(x, n, SC + slot);
    };

    // scale slots accumulate via atomicMax -> re-zero every call
    k_zero_u32<<<1, 128, 0, stream>>>(SC, 128);

    // ---------------- phase A: AdaIN-style renorm into X -------------------
    int s_c1 = sc++, s_s1 = sc++, s_s0 = sc++;
    absmax(content1, P1, s_c1);
    absmax(style1,   P1, s_s1);
    absmax(style0,   P1, s_s0);
    k_meanq<<<NB * CH, 256, 0, stream>>>(style1,   SC + s_s1, mean_s1, HW1);
    k_meanq<<<NB * CH, 256, 0, stream>>>(style0,   SC + s_s0, mean_s0, HW1);
    k_meanq<<<NB * CH, 256, 0, stream>>>(content1, SC + s_c1, mean_c1, HW1);
    k_q_submean<<<4096, 256, 0, stream>>>(content1, SC + s_c1, mean_c1, T1, P1, HW1);
    int s_x1p = sc++;
    absmax(T1, P1, s_x1p);
    k_quant_addmean<<<4096, 256, 0, stream>>>(T1, SC + s_x1p, mean_s1, X, P1, HW1);

    // ---------------- quantize modulation params + pack conv weights -------
    int s_w1 = sc++, s_b1 = sc++, s_w0 = sc++, s_b0 = sc++;
    absmax(weight1, 256, s_w1);
    absmax(bias1,   256, s_b1);
    absmax(weight0, 256, s_w0);
    absmax(bias0,   256, s_b0);
    k_quant_f32<<<1, 256, 0, stream>>>(weight1, SC + s_w1, qw1, 256);
    k_quant_f32<<<1, 256, 0, stream>>>(bias1,   SC + s_b1, qb1, 256);
    k_quant_f32<<<1, 256, 0, stream>>>(weight0, SC + s_w0, qw0, 256);
    k_quant_f32<<<1, 256, 0, stream>>>(bias0,   SC + s_b0, qb0, 256);

    const size_t W3N = (size_t)64 * 64 * 9;
    int s_wa = sc++, s_wb = sc++, s_wc = sc++, s_wd = sc++;
    absmax(w_d1_1, W3N, s_wa);
    absmax(w_d1_2, W3N, s_wb);
    absmax(w_d2_1, W3N, s_wc);
    absmax(w_d2_2, W3N, s_wd);
    k_pack3x3<<<144, 256, 0, stream>>>(w_d1_1, SC + s_wa, WP11, 64);
    k_pack3x3<<<144, 256, 0, stream>>>(w_d1_2, SC + s_wb, WP12, 64);
    k_pack3x3<<<144, 256, 0, stream>>>(w_d2_1, SC + s_wc, WP21, 64);
    k_pack3x3<<<144, 256, 0, stream>>>(w_d2_2, SC + s_wd, WP22, 64);

    // ---------------- residual block (filterMod) ---------------------------
    auto res_mod = [&](const float* Xin, float* OUT, const float* qw, const float* qb,
                       const half_t* WPa, const float* ba, const half_t* WPb,
                       const float* bb) {
        int s_qx = sc++;
        absmax(Xin, P1, s_qx);
        k_quant_both_nhwc<<<4096, 256, 0, stream>>>(Xin, SC + s_qx, Q32, Q16, P1, CH, HW1);
        k_conv3x3_wmma<<<4096, 128, 0, stream>>>(Q16, WPa, ba, T1, NB, HH, WW);
        int s_c = sc++;
        absmax(T1, P1, s_c);
        k_quant_f32<<<4096, 256, 0, stream>>>(T1, SC + s_c, T1, P1);       // x1 = fq(conv1)
        k_mod_mul<<<4096, 256, 0, stream>>>(T1, qw, T2, P1, HW1);
        int s_ta = sc++;
        absmax(T2, P1, s_ta);
        k_quant_f32<<<4096, 256, 0, stream>>>(T2, SC + s_ta, T2, P1);      // fq(qw*x1)
        k_mod_mul<<<4096, 256, 0, stream>>>(Q32, qb, T3, P1, HW1);
        int s_tb = sc++;
        absmax(T3, P1, s_tb);
        k_quant_f32<<<4096, 256, 0, stream>>>(T3, SC + s_tb, T3, P1);      // fq(qb*qx)
        k_add_relu<<<4096, 256, 0, stream>>>(T2, T3, T2, P1);              // relu(x2)
        int s_x3 = sc++;
        absmax(T2, P1, s_x3);
        k_quant_both_nhwc<<<4096, 256, 0, stream>>>(T2, SC + s_x3, T2, Q16, P1, CH, HW1);
        k_conv3x3_wmma<<<4096, 128, 0, stream>>>(Q16, WPb, bb, T3, NB, HH, WW);
        int s_c2 = sc++;
        absmax(T3, P1, s_c2);
        k_quant_f32<<<4096, 256, 0, stream>>>(T3, SC + s_c2, T3, P1);      // x4 = fq(conv2)
        k_mod_mul<<<4096, 256, 0, stream>>>(T3, qw, T1, P1, HW1);
        int s_t1 = sc++;
        absmax(T1, P1, s_t1);
        k_quant_f32<<<4096, 256, 0, stream>>>(T1, SC + s_t1, T1, P1);      // fq(qw*x4)
        k_mod_mul<<<4096, 256, 0, stream>>>(T2, qb, T3, P1, HW1);
        int s_t3 = sc++;
        absmax(T3, P1, s_t3);
        k_quant_f32<<<4096, 256, 0, stream>>>(T3, SC + s_t3, T3, P1);      // fq(qb*x3)
        k_add_relu<<<4096, 256, 0, stream>>>(T1, T3, T1, P1);              // relu(x5)
        int s_r5 = sc++;
        absmax(T1, P1, s_r5);
        k_quant_f32<<<4096, 256, 0, stream>>>(T1, SC + s_r5, T1, P1);      // fq(relu(x5))
        k_relu<<<4096, 256, 0, stream>>>(Xin, T3, P1);
        int s_rx = sc++;
        absmax(T3, P1, s_rx);
        k_quant_f32<<<4096, 256, 0, stream>>>(T3, SC + s_rx, T3, P1);      // fq(relu(x))
        k_add<<<4096, 256, 0, stream>>>(T3, T1, OUT, P1);
    };

    res_mod(X, RESO, qw1, qb1, WP11, b_d1_1, WP12, b_d1_2);

    // ---------------- second renorm + second residual block ----------------
    int s_r1 = sc++;
    absmax(RESO, P1, s_r1);
    k_meanq<<<NB * CH, 256, 0, stream>>>(RESO, SC + s_r1, mean_x2, HW1);
    k_q_submean<<<4096, 256, 0, stream>>>(RESO, SC + s_r1, mean_x2, T1, P1, HW1);
    int s_x3p = sc++;
    absmax(T1, P1, s_x3p);
    k_quant_addmean<<<4096, 256, 0, stream>>>(T1, SC + s_x3p, mean_s0, X, P1, HW1);

    res_mod(X, RESO, qw0, qb0, WP21, b_d2_1, WP22, b_d2_2);   // x4 -> RESO

    // ---------------- dec3 -------------------------------------------------
    int s_x4 = sc++;
    absmax(RESO, P1, s_x4);
    k_up2_q_f16<<<8192, 256, 0, stream>>>(RESO, SC + s_x4, U64_16, NB, CH, HH, WW);

    int s_g1 = sc++;
    absmax(w31, 32 * 2 * 9, s_g1);
    k_quant_f16<<<1, 256, 0, stream>>>(w31, SC + s_g1, W31Q, 576);
    k_gconv3x3_relu<<<8192, 256, 0, stream>>>(U64_16, W31Q, b31, G31, NB, 32, 256, 256);
    int s_g1o = sc++;
    absmax(G31, P2, s_g1o);
    k_quant_f16_nhwc<<<8192, 256, 0, stream>>>(G31, SC + s_g1o, G31_16, P2, 32, 256 * 256);

    int s_p32 = sc++;
    absmax(w32, 32 * 32, s_p32);
    k_pack1x1<<<4, 256, 0, stream>>>(w32, SC + s_p32, WP32);
    k_conv1x1_wmma32<<<16384, 64, 0, stream>>>(G31_16, WP32, b32, C32, NB, 256 * 256);
    int s_c32 = sc++;
    absmax(C32, P2, s_c32);
    k_up2_q_f16<<<16384, 256, 0, stream>>>(C32, SC + s_c32, U32_16, NB, 32, 256, 256);

    int s_g2 = sc++;
    absmax(w33, 16 * 2 * 9, s_g2);
    k_quant_f16<<<1, 256, 0, stream>>>(w33, SC + s_g2, W33Q, 288);
    k_gconv3x3_relu<<<16384, 256, 0, stream>>>(U32_16, W33Q, b33, G33, NB, 16, 512, 512);
    int s_g2o = sc++;
    absmax(G33, P3, s_g2o);
    k_quant_f16_nhwc<<<16384, 256, 0, stream>>>(G33, SC + s_g2o, G33_16, P3, 16, 512 * 512);

    int s_w34q = sc++;
    absmax(w34, 256, s_w34q);
    k_quant_f16<<<1, 256, 0, stream>>>(w34, SC + s_w34q, W34Q, 256);
    k_conv1x1_16_relu<<<16384, 256, 0, stream>>>(G33_16, W34Q, b34, C34, NB, 16, 512 * 512);
    int s_c34 = sc++;
    absmax(C34, P3, s_c34);
    k_quant_f16_nhwc<<<16384, 256, 0, stream>>>(C34, SC + s_c34, C34_16, P3, 16, 512 * 512);

    int s_w35q = sc++;
    absmax(w35, 3 * 16 * 81, s_w35q);
    k_pack9x9<<<82, 256, 0, stream>>>(w35, SC + s_w35q, WPX);
    k_conv9x9_wmma<<<16384, 128, 0, stream>>>(C34_16, WPX, b35, FIN, NB, 512, 512);

    int s_fin = sc++;
    absmax(FIN, PO, s_fin);
    k_quant_f32<<<8192, 256, 0, stream>>>(FIN, SC + s_fin, (float*)d_out, PO);
}
